// EmotionGRUCell_12644383719505
// MI455X (gfx1250) — matrix operand order repaired and checked by reading, not compiled, not run on hardware
//
#include <hip/hip_runtime.h>
#include <hip/hip_bf16.h>

typedef __attribute__((ext_vector_type(16))) _Float16 v16h;
typedef __attribute__((ext_vector_type(8)))  _Float16 h8;
typedef __attribute__((ext_vector_type(8)))  float    v8f;

#define BN 4096
#define TT 64
#define DD 256

// ---------------------------------------------------------------- utilities
__global__ void k_cast_f16(const float* __restrict__ src, _Float16* __restrict__ dst, int n) {
  int i = blockIdx.x * blockDim.x + threadIdx.x;
  if (i < n) dst[i] = (_Float16)src[i];
}

// ------------------------------------------------------------ gather/builds
// X_g (4096 x 768) = [sel(q0), sel(r0), U]
__global__ void k_build_xg(const float* __restrict__ q0, const float* __restrict__ r0,
                           const float* __restrict__ U, const float* __restrict__ qmask,
                           _Float16* __restrict__ dst) {
  int idx = blockIdx.x * blockDim.x + threadIdx.x;
  if (idx >= BN * 768) return;
  int b = idx / 768, c = idx - b * 768;
  float v;
  if (c < 256) {
    int qi = qmask[b * 2 + 1] > qmask[b * 2] ? 1 : 0;
    v = q0[((size_t)(b * 2 + qi)) * 256 + c];
  } else if (c < 512) {
    int qi = qmask[b * 2 + 1] > qmask[b * 2] ? 1 : 0;
    v = r0[((size_t)(b * 2 + qi)) * 256 + (c - 256)];
  } else {
    v = U[(size_t)b * 256 + (c - 512)];
  }
  dst[idx] = (_Float16)v;
}

// X (8192 x 512) = [S[b] broadcast over P, U[b]]
__global__ void k_build_xb(const float* __restrict__ S, const float* __restrict__ U,
                           _Float16* __restrict__ dst) {
  int idx = blockIdx.x * blockDim.x + threadIdx.x;
  if (idx >= BN * 2 * 512) return;
  int row = idx >> 9, c = idx & 511, b = row >> 1;
  float v = (c < 256) ? S[(size_t)b * 256 + c] : U[(size_t)b * 256 + (c - 256)];
  dst[idx] = (_Float16)v;
}

// X_pl (8192 x 512) = [sel(qs)[b] broadcast over P, U[b]]
__global__ void k_build_xsel(const float* __restrict__ qs, const float* __restrict__ qmask,
                             const float* __restrict__ U, _Float16* __restrict__ dst) {
  int idx = blockIdx.x * blockDim.x + threadIdx.x;
  if (idx >= BN * 2 * 512) return;
  int row = idx >> 9, c = idx & 511, b = row >> 1;
  float v;
  if (c < 256) {
    int qi = qmask[b * 2 + 1] > qmask[b * 2] ? 1 : 0;
    v = qs[((size_t)(b * 2 + qi)) * 256 + c];
  } else {
    v = U[(size_t)b * 256 + (c - 256)];
  }
  dst[idx] = (_Float16)v;
}

// X_e (4096 x 768) = [U, sel(q_), g_]
__global__ void k_build_xe(const float* __restrict__ U, const float* __restrict__ q_,
                           const float* __restrict__ qmask, const float* __restrict__ g_,
                           _Float16* __restrict__ dst) {
  int idx = blockIdx.x * blockDim.x + threadIdx.x;
  if (idx >= BN * 768) return;
  int b = idx / 768, c = idx - b * 768;
  float v;
  if (c < 256) {
    v = U[(size_t)b * 256 + c];
  } else if (c < 512) {
    int qi = qmask[b * 2 + 1] > qmask[b * 2] ? 1 : 0;
    v = q_[((size_t)(b * 2 + qi)) * 256 + (c - 256)];
  } else {
    v = g_[(size_t)b * 256 + (c - 512)];
  }
  dst[idx] = (_Float16)v;
}

// --------------------------------------------------------------- WMMA GEMM
// C[M,N] = A[M,K] @ W[N,K]^T + bias[N]   (A,W f16; C,bias f32)
// block = 256 threads = 8 waves; each wave computes a 16(M) x 64(N) strip
// (4 f32 accumulators). All A/B fragment loads for a K-block are pinned
// above the 4 WMMAs with a sched_barrier so the B frags stay in distinct
// registers and the WMMAs only take partial s_wait_loadcnt (overlap loads
// with matrix issue). ~80 live VGPRs: no spills.
// grid = (N/64, M/128). K % 32 == 0, M % 128 == 0, N % 64 == 0.
union FragU { v16h v; h8 h[2]; };

__global__ void __launch_bounds__(256)
k_wmma_gemm(const _Float16* __restrict__ A, const _Float16* __restrict__ W,
            const float* __restrict__ bias, float* __restrict__ C,
            int M, int N, int K) {
  int wave = threadIdx.x >> 5;
  int lane = threadIdx.x & 31;
  int hs   = lane >> 4;      // half-wave select
  int l15  = lane & 15;
  int m_base = blockIdx.y * 128 + wave * 16;
  int n_base = blockIdx.x * 64;

  // A frag per ISA layout: lane holds row m_base+l15,
  // K = {a_off..a_off+7, a_off+16..a_off+23}, a_off = hs ? 8 : 0.
  const _Float16* Arow = A + (size_t)(m_base + l15) * K + (hs ? 8 : 0);
  // B frag: lane holds column n (= weight row n), K = {b_off..b_off+7, b_off+16..b_off+23},
  // b_off = hs ? 8 : 0.
  const int b_off = hs ? 8 : 0;
  const _Float16* Wrow0 = W + (size_t)(n_base + l15) * K + b_off;
  const size_t wstride = (size_t)16 * K;

  v8f zero = {0.f, 0.f, 0.f, 0.f, 0.f, 0.f, 0.f, 0.f};
  v8f acc[4];
#pragma unroll
  for (int j = 0; j < 4; ++j) acc[j] = zero;

  for (int kb = 0; kb < K; kb += 32) {
    FragU a;
    a.h[0] = *(const h8*)(Arow + kb);
    a.h[1] = *(const h8*)(Arow + kb + 16);
    FragU b[4];
#pragma unroll
    for (int j = 0; j < 4; ++j) {
      const _Float16* p = Wrow0 + (size_t)j * wstride + kb;
      b[j].h[0] = *(const h8*)(p);
      b[j].h[1] = *(const h8*)(p + 16);
    }
    if (kb + 32 < K) __builtin_prefetch(Arow + kb + 32, 0, 1);
#if __has_builtin(__builtin_amdgcn_sched_barrier)
    // keep the 10 fragment loads above the WMMA group: forces distinct B
    // registers and partial s_wait_loadcnt instead of load->wait0->wmma
    __builtin_amdgcn_sched_barrier(0);
#endif
#pragma unroll
    for (int j = 0; j < 4; ++j) {
      acc[j] = __builtin_amdgcn_wmma_f32_16x16x32_f16(
          false, a.v, false, b[j].v, (short)0, acc[j], false, false);
    }
  }

#pragma unroll
  for (int j = 0; j < 4; ++j) {
    int col = n_base + j * 16 + l15;
    float bv = bias ? bias[col] : 0.f;
    float* Cp = C + (size_t)(m_base + (hs ? 8 : 0)) * N + col;
#pragma unroll
    for (int i = 0; i < 8; ++i) Cp[(size_t)i * N] = acc[j][i] + bv;  // row m_base+i (+8 hi half)
  }
}

// ---------------------------------------------------- attention (fused pass)
// scale[t] = g_hist[t,b,:]·att_w ; alpha = softmax_t ; c_[b] = sum_t alpha*g_hist[t,b]
// one wave per batch b; 8 waves per block. Single HBM pass over g_hist
// (second loop re-reads the 64KB/batch working set from L0/L2).
__global__ void k_attention(const float* __restrict__ g_hist, const float* __restrict__ att_w,
                            float* __restrict__ c_, float* __restrict__ alpha_out) {
  __shared__ float s_alpha[8][TT];
  int wave = threadIdx.x >> 5;
  int lane = threadIdx.x & 31;
  int b = blockIdx.x * 8 + wave;

  float w[8];
#pragma unroll
  for (int j = 0; j < 8; ++j) w[j] = att_w[lane * 8 + j];

  for (int t = 0; t < TT; ++t) {
    const float* gp = g_hist + ((size_t)t * BN + b) * DD + lane * 8;
    float dot = 0.f;
#pragma unroll
    for (int j = 0; j < 8; ++j) dot += gp[j] * w[j];
#pragma unroll
    for (int m = 16; m >= 1; m >>= 1) dot += __shfl_xor(dot, m, 32);
    if (lane == 0) s_alpha[wave][t] = dot;
  }
  __syncthreads();

  float v0 = s_alpha[wave][lane], v1 = s_alpha[wave][lane + 32];
  float mx = fmaxf(v0, v1);
#pragma unroll
  for (int m = 16; m >= 1; m >>= 1) mx = fmaxf(mx, __shfl_xor(mx, m, 32));
  float e0 = __expf(v0 - mx), e1 = __expf(v1 - mx);
  float s = e0 + e1;
#pragma unroll
  for (int m = 16; m >= 1; m >>= 1) s += __shfl_xor(s, m, 32);
  float inv = 1.f / s;
  float a0 = e0 * inv, a1 = e1 * inv;
  alpha_out[(size_t)b * TT + lane] = a0;
  alpha_out[(size_t)b * TT + lane + 32] = a1;
  s_alpha[wave][lane] = a0;
  s_alpha[wave][lane + 32] = a1;
  __syncthreads();

  float acc[8];
#pragma unroll
  for (int j = 0; j < 8; ++j) acc[j] = 0.f;
  for (int t = 0; t < TT; ++t) {
    float a = s_alpha[wave][t];
    const float* gp = g_hist + ((size_t)t * BN + b) * DD + lane * 8;
#pragma unroll
    for (int j = 0; j < 8; ++j) acc[j] += a * gp[j];
  }
  float* cp = c_ + (size_t)b * DD + lane * 8;
#pragma unroll
  for (int j = 0; j < 8; ++j) cp[j] = acc[j];
}

// ---------------------------------------------------------------- GRU gate
// mode 0: out = gru ; mode 1: out = gru*qmask[row] ; mode 2: out = gru*qm + qs*(1-qm)
__global__ void k_gru_gate(const float* __restrict__ gi, const float* __restrict__ gh,
                           const float* __restrict__ h, const float* __restrict__ qmask,
                           const float* __restrict__ qs, float* __restrict__ out,
                           int M, int mode) {
  int idx = blockIdx.x * blockDim.x + threadIdx.x;
  if (idx >= M * DD) return;
  int row = idx >> 8;
  int d = idx & 255;
  size_t o = (size_t)row * 768;
  float r = 1.f / (1.f + __expf(-(gi[o + d] + gh[o + d])));
  float z = 1.f / (1.f + __expf(-(gi[o + 256 + d] + gh[o + 256 + d])));
  float n = tanhf(gi[o + 512 + d] + r * gh[o + 512 + d]);
  float res = (1.f - z) * n + z * h[idx];
  if (mode == 1) {
    res *= qmask[row];
  } else if (mode == 2) {
    float qm = qmask[row];
    res = res * qm + qs[idx] * (1.f - qm);
  }
  out[idx] = res;
}

// ------------------------------------------------------------------- driver
extern "C" void kernel_launch(void* const* d_in, const int* in_sizes, int n_in,
                              void* d_out, int out_size, void* d_ws, size_t ws_size,
                              hipStream_t stream) {
  (void)in_sizes; (void)n_in; (void)out_size; (void)ws_size;
  const float* U      = (const float*)d_in[0];
  const float* qmask  = (const float*)d_in[1];
  const float* g_hist = (const float*)d_in[2];
  const float* q0     = (const float*)d_in[3];
  const float* r0     = (const float*)d_in[4];
  const float* e0     = (const float*)d_in[5];
  const float* g_wih  = (const float*)d_in[6];
  const float* g_whh  = (const float*)d_in[7];
  const float* g_bih  = (const float*)d_in[8];
  const float* g_bhh  = (const float*)d_in[9];
  const float* p_wih  = (const float*)d_in[10];
  const float* p_whh  = (const float*)d_in[11];
  const float* p_bih  = (const float*)d_in[12];
  const float* p_bhh  = (const float*)d_in[13];
  const float* pl_wih = (const float*)d_in[14];
  const float* pl_whh = (const float*)d_in[15];
  const float* pl_bih = (const float*)d_in[16];
  const float* pl_bhh = (const float*)d_in[17];
  const float* r_wih  = (const float*)d_in[18];
  const float* r_whh  = (const float*)d_in[19];
  const float* r_bih  = (const float*)d_in[20];
  const float* r_bhh  = (const float*)d_in[21];
  const float* e_wih  = (const float*)d_in[22];
  const float* e_whh  = (const float*)d_in[23];
  const float* e_bih  = (const float*)d_in[24];
  const float* e_bhh  = (const float*)d_in[25];
  const float* att_w  = (const float*)d_in[26];

  float* out       = (float*)d_out;
  float* out_g     = out;                 // 4096*256
  float* out_q     = out + 1048576;       // 4096*2*256
  float* out_r     = out + 3145728;       // 4096*2*256
  float* out_e     = out + 5242880;       // 4096*256
  float* out_alpha = out + 6291456;       // 4096*64

  // workspace layout
  size_t off = 0;
  char* ws = (char*)d_ws;
  auto allocH = [&](size_t elems) { _Float16* p = (_Float16*)(ws + off);
                                    off = (off + elems * 2 + 255) & ~(size_t)255; return p; };
  auto allocF = [&](size_t elems) { float* p = (float*)(ws + off);
                                    off = (off + elems * 4 + 255) & ~(size_t)255; return p; };

  _Float16* w_g_wih  = allocH(768 * 768);
  _Float16* w_g_whh  = allocH(768 * 256);
  _Float16* w_p_wih  = allocH(768 * 512);
  _Float16* w_p_whh  = allocH(768 * 256);
  _Float16* w_pl_wih = allocH(768 * 512);
  _Float16* w_pl_whh = allocH(768 * 256);
  _Float16* w_r_wih  = allocH(768 * 512);
  _Float16* w_r_whh  = allocH(768 * 256);
  _Float16* w_e_wih  = allocH(768 * 768);
  _Float16* w_e_whh  = allocH(768 * 256);
  _Float16* Xbuf     = allocH(8192 * 512);   // >= 4096*768 as well
  _Float16* Hbuf     = allocH(8192 * 256);
  float*    gi       = allocF(8192 * 768);
  float*    gh       = allocF(8192 * 768);
  float*    c_       = allocF(4096 * 256);
  float*    qs       = allocF(8192 * 256);

  dim3 blk(256);
  auto g1 = [](int n) { return dim3((unsigned)((n + 255) / 256)); };

  // ---- weight casts (once per call) ----
  k_cast_f16<<<g1(768*768), blk, 0, stream>>>(g_wih,  w_g_wih,  768*768);
  k_cast_f16<<<g1(768*256), blk, 0, stream>>>(g_whh,  w_g_whh,  768*256);
  k_cast_f16<<<g1(768*512), blk, 0, stream>>>(p_wih,  w_p_wih,  768*512);
  k_cast_f16<<<g1(768*256), blk, 0, stream>>>(p_whh,  w_p_whh,  768*256);
  k_cast_f16<<<g1(768*512), blk, 0, stream>>>(pl_wih, w_pl_wih, 768*512);
  k_cast_f16<<<g1(768*256), blk, 0, stream>>>(pl_whh, w_pl_whh, 768*256);
  k_cast_f16<<<g1(768*512), blk, 0, stream>>>(r_wih,  w_r_wih,  768*512);
  k_cast_f16<<<g1(768*256), blk, 0, stream>>>(r_whh,  w_r_whh,  768*256);
  k_cast_f16<<<g1(768*768), blk, 0, stream>>>(e_wih,  w_e_wih,  768*768);
  k_cast_f16<<<g1(768*256), blk, 0, stream>>>(e_whh,  w_e_whh,  768*256);

  const float* hlast = g_hist + (size_t)(TT - 1) * BN * DD;

  // ---- g-cell ----
  k_cast_f16<<<g1(BN*DD), blk, 0, stream>>>(hlast, Hbuf, BN*DD);
  k_build_xg<<<g1(BN*768), blk, 0, stream>>>(q0, r0, U, qmask, Xbuf);
  k_wmma_gemm<<<dim3(12, 32), blk, 0, stream>>>(Xbuf, w_g_wih, g_bih, gi, BN, 768, 768);
  k_wmma_gemm<<<dim3(12, 32), blk, 0, stream>>>(Hbuf, w_g_whh, g_bhh, gh, BN, 768, 256);
  k_gru_gate<<<g1(BN*DD), blk, 0, stream>>>(gi, gh, hlast, nullptr, nullptr, out_g, BN, 0);

  // ---- attention (fused dot / softmax / weighted-sum over g_hist) ----
  k_attention<<<dim3(BN / 8), blk, 0, stream>>>(g_hist, att_w, c_, out_alpha);

  // ---- r-cell ----
  k_cast_f16<<<g1(2*BN*DD), blk, 0, stream>>>(r0, Hbuf, 2*BN*DD);
  k_build_xb<<<g1(2*BN*512), blk, 0, stream>>>(c_, U, Xbuf);
  k_wmma_gemm<<<dim3(12, 64), blk, 0, stream>>>(Xbuf, w_r_wih, r_bih, gi, 2*BN, 768, 512);
  k_wmma_gemm<<<dim3(12, 64), blk, 0, stream>>>(Hbuf, w_r_whh, r_bhh, gh, 2*BN, 768, 256);
  k_gru_gate<<<g1(2*BN*DD), blk, 0, stream>>>(gi, gh, r0, qmask, nullptr, out_r, 2*BN, 1);

  // ---- p-cell ----
  k_cast_f16<<<g1(2*BN*DD), blk, 0, stream>>>(q0, Hbuf, 2*BN*DD);
  k_build_xb<<<g1(2*BN*512), blk, 0, stream>>>(out_g, U, Xbuf);
  k_wmma_gemm<<<dim3(12, 64), blk, 0, stream>>>(Xbuf, w_p_wih, p_bih, gi, 2*BN, 768, 512);
  k_wmma_gemm<<<dim3(12, 64), blk, 0, stream>>>(Hbuf, w_p_whh, p_bhh, gh, 2*BN, 768, 256);
  k_gru_gate<<<g1(2*BN*DD), blk, 0, stream>>>(gi, gh, q0, nullptr, nullptr, qs, 2*BN, 0);

  // ---- pl-cell (depends on qs) ----
  k_build_xsel<<<g1(2*BN*512), blk, 0, stream>>>(qs, qmask, U, Xbuf);
  k_wmma_gemm<<<dim3(12, 64), blk, 0, stream>>>(Xbuf, w_pl_wih, pl_bih, gi, 2*BN, 768, 512);
  k_wmma_gemm<<<dim3(12, 64), blk, 0, stream>>>(Hbuf, w_pl_whh, pl_bhh, gh, 2*BN, 768, 256);
  k_gru_gate<<<g1(2*BN*DD), blk, 0, stream>>>(gi, gh, q0, qmask, qs, out_q, 2*BN, 2);

  // ---- e-cell ----
  k_cast_f16<<<g1(BN*DD), blk, 0, stream>>>(e0, Hbuf, BN*DD);
  k_build_xe<<<g1(BN*768), blk, 0, stream>>>(U, out_q, qmask, out_g, Xbuf);
  k_wmma_gemm<<<dim3(12, 32), blk, 0, stream>>>(Xbuf, w_e_wih, e_bih, gi, BN, 768, 768);
  k_wmma_gemm<<<dim3(12, 32), blk, 0, stream>>>(Hbuf, w_e_whh, e_bhh, gh, BN, 768, 256);
  k_gru_gate<<<g1(BN*DD), blk, 0, stream>>>(gi, gh, e0, nullptr, nullptr, out_e, BN, 0);
}
